// Attention_22711787061815
// MI455X (gfx1250) — matrix operand order repaired and checked by reading, not compiled, not run on hardware
//
#include <hip/hip_runtime.h>
#include <hip/hip_bf16.h>

// ---- CDNA5 wave32 WMMA types ----
typedef __attribute__((ext_vector_type(16))) __bf16 v16bf;
typedef __attribute__((ext_vector_type(8)))  float  v8f;

// ---- TDM availability (device pass only; host pass falls back) ----
#if defined(__has_builtin)
#  if __has_builtin(__builtin_amdgcn_tensor_load_to_lds) && \
      __has_builtin(__builtin_amdgcn_tensor_store_from_lds) && \
      __has_builtin(__builtin_amdgcn_s_wait_tensorcnt)
#    define HAVE_TDM 1
#  endif
#endif
#ifndef HAVE_TDM
#  define HAVE_TDM 0
#endif

#if HAVE_TDM
typedef __attribute__((ext_vector_type(4))) unsigned v4u;
typedef __attribute__((ext_vector_type(8))) int      v8i;
typedef __attribute__((ext_vector_type(4))) int      v4i;

// D# for one contiguous row of 9216 bf16 (18432 B): 2D tensor, tile == tensor.
// group1: data_size=1(2B)<<16 ; tensor_dim0=9216 ; tensor_dim1=1 ; tile_dim0=9216 ;
//         tile_dim1=1 ; tensor_dim0_stride=9216
__device__ __forceinline__ v8i tdm_g1_row(void) {
    v8i g1 = { 0x10000, 0x24000000, 0x00010000, 0x24000000, 1, 9216, 0, 0 };
    return g1;
}
__device__ __forceinline__ v4u tdm_g0(unsigned long long gaddr, const void* ldsp) {
    const unsigned lds = (unsigned)(unsigned long long)ldsp;  // generic->LDS offset (addr[31:0])
    v4u g0 = { 1u,                                            // count=1, user descriptor
               lds,                                           // lds_addr
               (unsigned)gaddr,                               // global_addr[31:0]
               (unsigned)((gaddr >> 32) & 0x1FFFFFFu) | 0x80000000u }; // [56:32] | type=2
    return g0;
}
__device__ __forceinline__ void tdm_load_row(const void* gsrc, void* ldst) {
    v4i z4 = { 0, 0, 0, 0 };
    v8i z8 = { 0, 0, 0, 0, 0, 0, 0, 0 };
    __builtin_amdgcn_tensor_load_to_lds(tdm_g0((unsigned long long)gsrc, ldst),
                                        tdm_g1_row(), z4, z4, z8, 0);
}
__device__ __forceinline__ void tdm_store_row(void* gdst, const void* lsrc) {
    v4i z4 = { 0, 0, 0, 0 };
    v8i z8 = { 0, 0, 0, 0, 0, 0, 0, 0 };
    __builtin_amdgcn_tensor_store_from_lds(tdm_g0((unsigned long long)gdst, lsrc),
                                           tdm_g1_row(), z4, z4, z8, 0);
}
#endif

// ---------------------------------------------------------------------------
// Kernel 0: one-time f32 -> bf16 conversion (x, w_vkq, w_out)
// ---------------------------------------------------------------------------
__global__ void k_f32_to_bf16(const float* __restrict__ in, __bf16* __restrict__ out, int n)
{
    int i = blockIdx.x * blockDim.x + threadIdx.x;
    const int stride = gridDim.x * blockDim.x;
    for (; i < n; i += stride) out[i] = (__bf16)in[i];
}

// ---------------------------------------------------------------------------
// Kernel 1: vqk projection.  Out[o,p] = sum_c W[o,c] * X[b,c,p] + bias[o]
// M=1536, K=64 (2 wmma steps), N=9216. One wave per 16x16 tile.
// ---------------------------------------------------------------------------
__global__ __launch_bounds__(128) void k_vkq_proj(
    const __bf16* __restrict__ X, const __bf16* __restrict__ W,
    const float* __restrict__ bias,
    __bf16* __restrict__ V, __bf16* __restrict__ Q, __bf16* __restrict__ K)
{
    const int lane = threadIdx.x & 31;
    const int wave = threadIdx.x >> 5;
    const int lh = lane >> 4, ln = lane & 15;
    const int b  = blockIdx.z;
    const int m0 = blockIdx.y * 16;
    const int n0 = (blockIdx.x * 4 + wave) * 16;
    const __bf16* Xb = X + (size_t)b * 64 * 9216;

    v8f acc = {};
#pragma unroll
    for (int k0 = 0; k0 < 64; k0 += 32) {
        v16bf a, bb;
        const __bf16* ap = W + (m0 + ln) * 64 + k0 + lh * 8;
#pragma unroll
        for (int j = 0; j < 8; ++j) { a[j] = ap[j]; a[j + 8] = ap[16 + j]; }
        const __bf16* bp = Xb + (size_t)(k0 + lh * 8) * 9216 + n0 + ln;
        __builtin_prefetch(bp + (size_t)32 * 9216, 0, 0);
#pragma unroll
        for (int j = 0; j < 8; ++j) { bb[j] = bp[(size_t)j * 9216]; bb[j + 8] = bp[(size_t)(16 + j) * 9216]; }
        acc = __builtin_amdgcn_wmma_f32_16x16x32_bf16(false, a, false, bb,
                                                      (short)0, acc, false, false);
    }

    const int p = n0 + ln;
#pragma unroll
    for (int r = 0; r < 8; ++r) {
        const int m = m0 + r + lh * 8;
        const float val = acc[r] + bias[m];
        const int s  = m >> 9;          // 0:v 1:q 2:k (uniform per r)
        const int h  = (m >> 6) & 7;
        const int cc = m & 63;
        const size_t idx = ((size_t)((b * 8 + h) * 64 + cc)) * 9216 + p;
        const __bf16 bv = (__bf16)val;
        if (s == 0)      V[idx] = bv;
        else if (s == 1) Q[idx] = bv;
        else             K[idx] = bv;
    }
}

// ---------------------------------------------------------------------------
// Kernel 2: per-(b,h,c) attention, LDS resident, TDM in/out.
//   S = V*K^T/9216 ; row softmax ; O = softmax(S)*Q^T
// 192 threads (6 waves); wave w owns rows [16w,16w+16).
// sK is reused (after stage 1) for bf16 softmaxed scores; sV (after stage 1)
// is reused as the output staging buffer for the TDM store.
// ---------------------------------------------------------------------------
__global__ __launch_bounds__(192) void k_attention(
    const __bf16* __restrict__ V, const __bf16* __restrict__ Q,
    const __bf16* __restrict__ K, __bf16* __restrict__ AttnOut)
{
    __shared__ __bf16 sV[96 * 96];   // V, later: output staging
    __shared__ __bf16 sK[96 * 96];   // K, later: bf16 softmaxed scores
    __shared__ __bf16 sQ[96 * 96];
    __shared__ float  sS[96 * 96];   // fp32 scores

    const int tid  = threadIdx.x;
    const int lane = tid & 31;
    const int wave = tid >> 5;
    const int lh = lane >> 4, ln = lane & 15;

    const int slice = blockIdx.x;                 // b*512 + h*64 + cc
    const size_t base = (size_t)slice * 9216;

#if HAVE_TDM
    if (wave == 0) {
        tdm_load_row(V + base, sV);
        tdm_load_row(K + base, sK);
        tdm_load_row(Q + base, sQ);
        __builtin_amdgcn_s_wait_tensorcnt(0);
    }
#else
    {
        const uint4* gv = (const uint4*)(V + base);
        const uint4* gq = (const uint4*)(Q + base);
        const uint4* gk = (const uint4*)(K + base);
        uint4* lv = (uint4*)sV; uint4* lq = (uint4*)sQ; uint4* lk = (uint4*)sK;
        for (int i = tid; i < 1152; i += 192) { lv[i] = gv[i]; lk[i] = gk[i]; lq[i] = gq[i]; }
    }
#endif
    __syncthreads();

    const int x0 = wave * 16;

    // ---- stage 1: S[x,z] = (1/9216) * sum_y V[x,y]*K[z,y] ----
    for (int zt = 0; zt < 6; ++zt) {
        v8f acc = {};
#pragma unroll
        for (int y0 = 0; y0 < 96; y0 += 32) {
            v16bf a, bb;
            const __bf16* ap = sV + (x0 + ln) * 96 + y0 + lh * 8;
#pragma unroll
            for (int j = 0; j < 8; ++j) { a[j] = ap[j]; a[j + 8] = ap[16 + j]; }
            const __bf16* bp = sK + (zt * 16 + ln) * 96 + y0 + lh * 8;
#pragma unroll
            for (int j = 0; j < 8; ++j) { bb[j] = bp[j]; bb[j + 8] = bp[16 + j]; }
            acc = __builtin_amdgcn_wmma_f32_16x16x32_bf16(false, a, false, bb,
                                                          (short)0, acc, false, false);
        }
        const float scale = 1.0f / 9216.0f;
#pragma unroll
        for (int r = 0; r < 8; ++r)
            sS[(x0 + r + lh * 8) * 96 + zt * 16 + ln] = acc[r] * scale;
    }
    __syncthreads();   // also: last read of sV / sK data

    // ---- stage 2: row softmax; normalized bf16 result goes into sK ----
    if (tid < 96) {
        const float* row = sS + tid * 96;
        __bf16* rowb = sK + tid * 96;
        float m = -1e30f;
        for (int j = 0; j < 96; ++j) m = fmaxf(m, row[j]);
        float s = 0.0f;
        float e[96];
        for (int j = 0; j < 96; ++j) { e[j] = __expf(row[j] - m); s += e[j]; }
        const float inv = 1.0f / s;
        for (int j = 0; j < 96; ++j) rowb[j] = (__bf16)(e[j] * inv);
    }
    __syncthreads();

    // ---- stage 3: O[x,z] = sum_y P[x,y]*Q[z,y]; stage into sV ----
    for (int zt = 0; zt < 6; ++zt) {
        v8f acc = {};
#pragma unroll
        for (int y0 = 0; y0 < 96; y0 += 32) {
            v16bf a, bb;
            const __bf16* ap = sK + (x0 + ln) * 96 + y0 + lh * 8;
#pragma unroll
            for (int j = 0; j < 8; ++j) { a[j] = ap[j]; a[j + 8] = ap[16 + j]; }
            const __bf16* bp = sQ + (zt * 16 + ln) * 96 + y0 + lh * 8;
#pragma unroll
            for (int j = 0; j < 8; ++j) { bb[j] = bp[j]; bb[j + 8] = bp[16 + j]; }
            acc = __builtin_amdgcn_wmma_f32_16x16x32_bf16(false, a, false, bb,
                                                          (short)0, acc, false, false);
        }
#pragma unroll
        for (int r = 0; r < 8; ++r)
            sV[(x0 + r + lh * 8) * 96 + zt * 16 + ln] = (__bf16)acc[r];
    }
    __syncthreads();

    // write-back in (b, cc*8+h, x, z) layout for the final projection
    const int bb_ = slice >> 9;
    const int hh  = (slice >> 6) & 7;
    const int cc  = slice & 63;
    __bf16* Ob = AttnOut + ((size_t)(bb_ * 512 + cc * 8 + hh)) * 9216;

#if HAVE_TDM
    if (wave == 0) tdm_store_row(Ob, sV);   // S_ENDPGM implicitly drains TENSORcnt
#else
    {
        const uint4* lo = (const uint4*)sV;
        uint4* go = (uint4*)Ob;
        for (int i = tid; i < 1152; i += 192) go[i] = lo[i];
    }
#endif
}

// ---------------------------------------------------------------------------
// Kernel 3: output projection. Out[b,o,p] = sum_k Wout[o,k]*Attn[b,k,p] + bo[o]
// M=64, K=512 (16 wmma steps), N=9216.
// ---------------------------------------------------------------------------
__global__ __launch_bounds__(128) void k_out_proj(
    const __bf16* __restrict__ Attn, const __bf16* __restrict__ W,
    const float* __restrict__ bias, float* __restrict__ Out)
{
    const int lane = threadIdx.x & 31;
    const int wave = threadIdx.x >> 5;
    const int lh = lane >> 4, ln = lane & 15;
    const int b  = blockIdx.z;
    const int m0 = blockIdx.y * 16;
    const int n0 = (blockIdx.x * 4 + wave) * 16;
    const __bf16* Ab = Attn + (size_t)b * 512 * 9216;

    v8f acc = {};
    for (int k0 = 0; k0 < 512; k0 += 32) {
        v16bf a, bb;
        const __bf16* ap = W + (m0 + ln) * 512 + k0 + lh * 8;
#pragma unroll
        for (int j = 0; j < 8; ++j) { a[j] = ap[j]; a[j + 8] = ap[16 + j]; }
        const __bf16* bp = Ab + (size_t)(k0 + lh * 8) * 9216 + n0 + ln;
        __builtin_prefetch(bp + (size_t)32 * 9216, 0, 0);
#pragma unroll
        for (int j = 0; j < 8; ++j) { bb[j] = bp[(size_t)j * 9216]; bb[j + 8] = bp[(size_t)(16 + j) * 9216]; }
        acc = __builtin_amdgcn_wmma_f32_16x16x32_bf16(false, a, false, bb,
                                                      (short)0, acc, false, false);
    }

    const int p = n0 + ln;
    float* Ob = Out + (size_t)b * 64 * 9216;
#pragma unroll
    for (int r = 0; r < 8; ++r) {
        const int m = m0 + r + lh * 8;
        Ob[(size_t)m * 9216 + p] = acc[r] + bias[m];
    }
}

// ---------------------------------------------------------------------------
extern "C" void kernel_launch(void* const* d_in, const int* in_sizes, int n_in,
                              void* d_out, int out_size, void* d_ws, size_t ws_size,
                              hipStream_t stream)
{
    const float* x     = (const float*)d_in[0];   // (4,64,96,96)
    const float* w_vkq = (const float*)d_in[1];   // (1536,64)
    const float* b_vkq = (const float*)d_in[2];   // (1536,)
    const float* w_out = (const float*)d_in[3];   // (64,512)
    const float* b_out = (const float*)d_in[4];   // (64,)
    float* out = (float*)d_out;                   // (4,64,96,96)

    const size_t SLICES = 2048, PIX = 9216;
    __bf16* wsV    = (__bf16*)d_ws;               // 2048*9216
    __bf16* wsQ    = wsV + SLICES * PIX;
    __bf16* wsK    = wsQ + SLICES * PIX;
    __bf16* wsA    = wsK + SLICES * PIX;
    __bf16* wsXb   = wsA + SLICES * PIX;          // 4*64*9216
    __bf16* wsWvkq = wsXb + (size_t)4 * 64 * PIX; // 1536*64
    __bf16* wsWout = wsWvkq + 1536 * 64;          // 64*512

    // 0) one-time bf16 conversion of inputs/weights
    k_f32_to_bf16<<<512, 256, 0, stream>>>(x, wsXb, 4 * 64 * 9216);
    k_f32_to_bf16<<<96,  256, 0, stream>>>(w_vkq, wsWvkq, 1536 * 64);
    k_f32_to_bf16<<<32,  256, 0, stream>>>(w_out, wsWout, 64 * 512);

    // 1) vqk projection
    dim3 g1(144, 96, 4);
    k_vkq_proj<<<g1, 128, 0, stream>>>(wsXb, wsWvkq, b_vkq, wsV, wsQ, wsK);

    // 2) attention: one block per (b,h,c) slice
    k_attention<<<2048, 192, 0, stream>>>(wsV, wsQ, wsK, wsA);

    // 3) output projection
    dim3 g3(144, 4, 4);
    k_out_proj<<<g3, 128, 0, stream>>>(wsA, wsWout, b_out, out);
}
